// SpectralTransformer_34548716929434
// MI455X (gfx1250) — matrix operand order repaired and checked by reading, not compiled, not run on hardware
//
#include <hip/hip_runtime.h>
#include <hip/hip_bf16.h>
#include <math.h>

// ---------------- problem constants ----------------
#define BB 4
#define CC 64
#define HH 256
#define WW 256
#define NN 65536           // H*W
#define HEADS 8
#define DH 64
#define INNER 512
#define CM 256             // C*MULT

typedef __attribute__((ext_vector_type(16))) _Float16 v16h;
typedef __attribute__((ext_vector_type(8)))  float    v8f;
typedef __attribute__((ext_vector_type(2)))  float    v2f;

union V16HU { v16h v; _Float16 e[16]; uint4 q[2]; };
union V8F   { v8f  v; float e[8]; };
union V2F   { v2f  v; float e[2]; };

__device__ __forceinline__ v8f wmma_f32(v2f a, v2f b, v8f c) {
  return __builtin_amdgcn_wmma_f32_16x16x4_f32(false, a, false, b, (short)0, c, false, false);
}
__device__ __forceinline__ v8f wmma_f16(v16h a, v16h b, v8f c) {
  return __builtin_amdgcn_wmma_f32_16x16x32_f16(false, a, false, b, (short)0, c, false, false);
}

// Load a 16-half fragment as two aligned 16B LDS loads (ds_load_b128 x2).
__device__ __forceinline__ v16h ld_frag(const _Float16* p) {
  V16HU r;
  r.q[0] = *(const uint4*)(p);
  r.q[1] = *(const uint4*)(p + 8);
  return r.v;
}

// K permutation (within each 32-K block) so that each lane's A fragment is
// contiguous: layout order [0..7, 16..23, 8..15, 24..31]; lane hs=0 reads
// positions 0..15 (K 0..7,16..23), hs=1 reads 16..31 (K 8..15,24..31).
__device__ __forceinline__ int kperm(int k) {
  int r = k & 31;
  int pos = (r < 8) ? r : (r < 16 ? r + 8 : (r < 24 ? r - 8 : r));
  return (k & ~31) | pos;
}

__device__ __forceinline__ float gelu_exact(float x) {
  return 0.5f * x * (1.0f + erff(x * 0.70710678118654752440f));
}

// ---------------------------------------------------------------------------
// Kernel 1: LayerNorm over channels + per-workgroup partial Gram (f32 WMMA)
// grid = B*256 (256 pixels each), block = 256
// ---------------------------------------------------------------------------
__global__ __launch_bounds__(256) void k_ln_gram(
    const float* __restrict__ x, const float* __restrict__ gamma,
    const float* __restrict__ beta, float* __restrict__ xn,
    float* __restrict__ gpart)
{
  __shared__ float tile[64][260];   // row stride 260 dwords (bank step 4)
  int wg = blockIdx.x;
  int b  = wg >> 8;
  int g  = wg & 255;
  long base = (long)b * CC * NN + (long)g * 256;
  int t = threadIdx.x;               // one pixel per thread

  float s = 0.f, s2 = 0.f;
  for (int c = 0; c < 64; ++c) {
    float v = x[base + (long)c * NN + t];
    tile[c][t] = v; s += v; s2 += v * v;
  }
  float m  = s * (1.f / 64.f);
  float vv = s2 * (1.f / 64.f) - m * m;
  float rs = rsqrtf(vv + 1e-5f);
  for (int c = 0; c < 64; ++c) {
    float v = (tile[c][t] - m) * rs * gamma[c] + beta[c];
    tile[c][t] = v;
    xn[base + (long)c * NN + t] = v;
  }
  __syncthreads();

  int wv = t >> 5, lane = t & 31, lm = lane & 15, hs = lane >> 4;
  float* gp = gpart + (long)wg * 4096;
  for (int rep = 0; rep < 2; ++rep) {
    int tt = wv * 2 + rep;                  // 16 tiles of the 64x64 output
    int m0 = (tt >> 2) * 16, n0 = (tt & 3) * 16;
    v8f acc = {};
    for (int k0 = 0; k0 < 256; k0 += 4) {
      V2F a, bb;
      a.e[0]  = tile[m0 + lm][k0 + hs * 2 + 0];
      a.e[1]  = tile[m0 + lm][k0 + hs * 2 + 1];
      bb.e[0] = tile[n0 + lm][k0 + hs * 2 + 0];   // B = tile^T
      bb.e[1] = tile[n0 + lm][k0 + hs * 2 + 1];
      acc = wmma_f32(a.v, bb.v, acc);
    }
    V8F r; r.v = acc;
    for (int q = 0; q < 8; ++q)
      gp[(m0 + q + hs * 8) * 64 + n0 + lm] = r.e[q];
  }
}

// ---------------------------------------------------------------------------
// Kernel 2: reduce 256 partial Grams per batch (deterministic order)
// ---------------------------------------------------------------------------
__global__ __launch_bounds__(256) void k_gram_reduce(
    const float* __restrict__ gpart, float* __restrict__ G)
{
  int b = blockIdx.x, t = threadIdx.x;
  for (int i = t; i < 4096; i += 256) {
    const float* p = gpart + (long)b * 256 * 4096 + i;
    float s = 0.f;
    for (int g = 0; g < 256; ++g) s += p[(long)g * 4096];
    G[b * 4096 + i] = s;
  }
}

// ---------------------------------------------------------------------------
// Kernel 3: per (b,h): logits = (Wk G Wq^T)/(|k||q|) * rescale, softmax,
// write attn to output, and M2_h = attn_h @ Wv_h.   grid=B*HEADS, block=64
// ---------------------------------------------------------------------------
__global__ __launch_bounds__(64) void k_attn(
    const float* __restrict__ G, const float* __restrict__ Wq,
    const float* __restrict__ Wk, const float* __restrict__ Wv,
    const float* __restrict__ rescale, float* __restrict__ attn_out,
    float* __restrict__ M2)
{
  __shared__ float sG[64][64], sWq[64][64], sWk[64][64], sWv[64][64];
  __shared__ float T1[64][64], T2[64][64], S[64][64];
  __shared__ float nq[64];
  int b = blockIdx.x >> 3, h = blockIdx.x & 7;
  int t = threadIdx.x;                        // 0..63 (row id)

  for (int c = 0; c < 64; ++c) {
    sG [c][t] = G [b * 4096 + c * 64 + t];
    sWq[c][t] = Wq[(h * 64 + c) * 64 + t];
    sWk[c][t] = Wk[(h * 64 + c) * 64 + t];
    sWv[c][t] = Wv[(h * 64 + c) * 64 + t];
  }
  __syncthreads();

  for (int c2 = 0; c2 < 64; ++c2) {
    float a1 = 0.f, a2 = 0.f;
    for (int c1 = 0; c1 < 64; ++c1) {
      a1 += sWk[t][c1] * sG[c1][c2];
      a2 += sWq[t][c1] * sG[c1][c2];
    }
    T1[t][c2] = a1; T2[t][c2] = a2;
  }
  __syncthreads();

  float dk = 0.f, dq = 0.f;
  for (int c = 0; c < 64; ++c) { dk += T1[t][c] * sWk[t][c]; dq += T2[t][c] * sWq[t][c]; }
  float nk_t = fmaxf(sqrtf(fmaxf(dk, 0.f)), 1e-12f);
  nq[t] = fmaxf(sqrtf(fmaxf(dq, 0.f)), 1e-12f);
  for (int e = 0; e < 64; ++e) {
    float s = 0.f;
    for (int c = 0; c < 64; ++c) s += T1[t][c] * sWq[e][c];   // S = T1 Wq^T
    S[t][e] = s;
  }
  __syncthreads();

  float rsc = rescale[h];
  float inv_nk = 1.f / nk_t;
  float row[64];
  float mx = -3.4e38f;
  for (int e = 0; e < 64; ++e) {
    float l = S[t][e] * inv_nk * (1.f / nq[e]) * rsc;
    row[e] = l; mx = fmaxf(mx, l);
  }
  float sum = 0.f;
  for (int e = 0; e < 64; ++e) { float ex = __expf(row[e] - mx); row[e] = ex; sum += ex; }
  float inv = 1.f / sum;
  for (int e = 0; e < 64; ++e) {
    float a = row[e] * inv;
    S[t][e] = a;
    attn_out[((long)(b * 8 + h) * 64 + t) * 64 + e] = a;
  }
  __syncthreads();

  for (int c = 0; c < 64; ++c) {
    float s = 0.f;
    for (int e = 0; e < 64; ++e) s += S[t][e] * sWv[e][c];
    M2[(long)b * INNER * 64 + (h * 64 + t) * 64 + c] = s;
  }
}

// ---------------------------------------------------------------------------
// Kernel 4: P[b] = Wp (64x512) @ M2[b] (512x64).  grid=B, block=256
// ---------------------------------------------------------------------------
__global__ __launch_bounds__(256) void k_pmat(
    const float* __restrict__ Wp, const float* __restrict__ M2,
    float* __restrict__ P)
{
  int b = blockIdx.x, t = threadIdx.x;
  const float* m2 = M2 + (long)b * INNER * 64;
  for (int i = t; i < 4096; i += 256) {
    int c = i >> 6, cp = i & 63;
    float s = 0.f;
    for (int e = 0; e < INNER; ++e) s += Wp[c * INNER + e] * m2[e * 64 + cp];
    P[b * 4096 + i] = s;
  }
}

// ---------------------------------------------------------------------------
// Kernel 5: x1 = x + P@xn + bp  (f32 WMMA), then fused LayerNorm2 -> xn2 (f16)
// grid = B*1024 (64 pixels each), block = 256
// ---------------------------------------------------------------------------
__global__ __launch_bounds__(256) void k_apply_ln2(
    const float* __restrict__ x, const float* __restrict__ xn,
    const float* __restrict__ P, const float* __restrict__ bp,
    const float* __restrict__ gamma, const float* __restrict__ beta,
    float* __restrict__ x1, _Float16* __restrict__ xn2)
{
  __shared__ float sXn[64][68];
  __shared__ float sP [64][68];
  __shared__ float sX1[64][68];
  __shared__ float smean[64], srs[64];
  int wg = blockIdx.x;
  int b = wg >> 10, g = wg & 1023;
  long base = (long)b * CC * NN + (long)g * 64;
  int t = threadIdx.x;

  for (int i = t; i < 4096; i += 256) {
    int c = i >> 6, p = i & 63;
    sXn[c][p] = xn[base + (long)c * NN + p];
    sP [c][p] = P[b * 4096 + i];
  }
  __syncthreads();

  int wv = t >> 5, lane = t & 31, lm = lane & 15, hs = lane >> 4;
  for (int rep = 0; rep < 2; ++rep) {
    int tt = wv * 2 + rep;
    int m0 = (tt >> 2) * 16, n0 = (tt & 3) * 16;
    v8f acc = {};
    for (int k0 = 0; k0 < 64; k0 += 4) {
      V2F a, bb;
      a.e[0]  = sP [m0 + lm][k0 + hs * 2 + 0];
      a.e[1]  = sP [m0 + lm][k0 + hs * 2 + 1];
      bb.e[0] = sXn[k0 + hs * 2 + 0][n0 + lm];
      bb.e[1] = sXn[k0 + hs * 2 + 1][n0 + lm];
      acc = wmma_f32(a.v, bb.v, acc);
    }
    V8F r; r.v = acc;
    for (int q = 0; q < 8; ++q) {
      int mm = m0 + q + hs * 8, nn = n0 + lm;
      sX1[mm][nn] = r.e[q] + bp[mm] + x[base + (long)mm * NN + nn];
    }
  }
  __syncthreads();

  if (t < 64) {               // per-pixel LN stats over 64 channels
    float s = 0.f, s2 = 0.f;
    for (int c = 0; c < 64; ++c) { float v = sX1[c][t]; s += v; s2 += v * v; }
    float m = s * (1.f / 64.f);
    smean[t] = m;
    srs[t] = rsqrtf(s2 * (1.f / 64.f) - m * m + 1e-5f);
  }
  __syncthreads();

  for (int i = t; i < 4096; i += 256) {
    int c = i >> 6, p = i & 63;
    float v = sX1[c][p];
    x1[base + (long)c * NN + p] = v;
    xn2[base + (long)c * NN + p] =
        (_Float16)((v - smean[p]) * srs[p] * gamma[c] + beta[c]);
  }
}

// ---------------------------------------------------------------------------
// Kernel 6: FFN1  y1 = gelu(W1 (256x64) @ xn2)   (f16 WMMA 16x16x32)
// grid = B*1024, block = 256
// A: K-permuted contiguous layout; B: transposed tile -> 2x b128 per fragment
// ---------------------------------------------------------------------------
__global__ __launch_bounds__(256) void k_ffn1(
    const _Float16* __restrict__ xn2, const float* __restrict__ w1,
    _Float16* __restrict__ y1)
{
  __shared__ _Float16 sW [256][72];   // W1, K-permuted (row 144B, 16B aligned)
  __shared__ _Float16 sXT[64][72];    // xn2 tile transposed: [pixel][K]
  int wg = blockIdx.x;
  int b = wg >> 10, g = wg & 1023;
  long base  = (long)b * CC * NN + (long)g * 64;
  long obase = (long)b * CM * NN + (long)g * 64;
  int t = threadIdx.x;

  // weights: 256x64 f32 -> f16, 16B global loads, permuted store
  const float4* w4 = (const float4*)w1;
  for (int ch = t; ch < 4096; ch += 256) {
    float4 v = w4[ch];
    int m = ch >> 4, k = (ch & 15) * 4;
    int kp = kperm(k);                       // 4 consecutive positions
    sW[m][kp + 0] = (_Float16)v.x;
    sW[m][kp + 1] = (_Float16)v.y;
    sW[m][kp + 2] = (_Float16)v.z;
    sW[m][kp + 3] = (_Float16)v.w;
  }
  // activations: read 8 halves along pixels, scatter-transpose into LDS
  for (int ch = t; ch < 512; ch += 256) {
    int c = ch >> 3, pc = (ch & 7) * 8;
    union { uint4 q; _Float16 h[8]; } u;
    u.q = ((const uint4*)(xn2 + base + (long)c * NN))[ch & 7];
    #pragma unroll
    for (int j = 0; j < 8; ++j) sXT[pc + j][kperm(c)] = u.h[j];
  }
  __syncthreads();

  int wv = t >> 5, lane = t & 31, lm = lane & 15, hs = lane >> 4;
  v8f z = {};
  v8f acc[2][4];
  #pragma unroll
  for (int m2 = 0; m2 < 2; ++m2)
    #pragma unroll
    for (int n = 0; n < 4; ++n) acc[m2][n] = z;

  #pragma unroll
  for (int k0 = 0; k0 < 64; k0 += 32) {
    v16h a0 = ld_frag(&sW[(wv * 2 + 0) * 16 + lm][k0 + hs * 16]);
    v16h a1 = ld_frag(&sW[(wv * 2 + 1) * 16 + lm][k0 + hs * 16]);
    #pragma unroll
    for (int n = 0; n < 4; ++n) {
      v16h bb = ld_frag(&sXT[n * 16 + lm][k0 + hs * 16]);
      acc[0][n] = wmma_f16(a0, bb, acc[0][n]);
      acc[1][n] = wmma_f16(a1, bb, acc[1][n]);
    }
  }

  #pragma unroll
  for (int m2 = 0; m2 < 2; ++m2)
    #pragma unroll
    for (int n = 0; n < 4; ++n) {
      V8F r; r.v = acc[m2][n];
      int m0 = (wv * 2 + m2) * 16, n0 = n * 16;
      #pragma unroll
      for (int q = 0; q < 8; ++q) {
        int mm = m0 + q + hs * 8, nn = n0 + lm;
        y1[obase + (long)mm * NN + nn] = (_Float16)gelu_exact(r.e[q]);
      }
    }
}

// ---------------------------------------------------------------------------
// Kernel 7: depthwise 3x3 'SAME' + gelu.  grid = B*CM*N/256, block = 256
// ---------------------------------------------------------------------------
__global__ __launch_bounds__(256) void k_dw(
    const _Float16* __restrict__ y1, const float* __restrict__ wd,
    _Float16* __restrict__ yd)
{
  long idx = (long)blockIdx.x * 256 + threadIdx.x;   // over B*CM*N
  int p = (int)(idx & (NN - 1));
  long rest = idx >> 16;
  int ch = (int)(rest & 255);
  int yy = p >> 8, xx = p & 255;
  const _Float16* src = y1 + (rest << 16);
  const float* w = wd + ch * 9;
  float acc = 0.f;
  #pragma unroll
  for (int ky = 0; ky < 3; ++ky) {
    int iy = yy + ky - 1;
    if (iy < 0 || iy > 255) continue;
    #pragma unroll
    for (int kx = 0; kx < 3; ++kx) {
      int ix = xx + kx - 1;
      if (ix < 0 || ix > 255) continue;
      acc += w[ky * 3 + kx] * (float)src[iy * 256 + ix];
    }
  }
  yd[idx] = (_Float16)gelu_exact(acc);
}

// ---------------------------------------------------------------------------
// Kernel 8: out = W2 (64x256) @ yd + x1   (f16 WMMA), grid = B*1024, block 256
// ---------------------------------------------------------------------------
__global__ __launch_bounds__(256) void k_ffn2(
    const _Float16* __restrict__ yd, const float* __restrict__ w2,
    const float* __restrict__ x1, float* __restrict__ out)
{
  __shared__ _Float16 sW [64][264];   // W2, K-permuted (row 528B, 16B aligned)
  __shared__ _Float16 sYT[64][264];   // yd tile transposed: [pixel][K]
  int wg = blockIdx.x;
  int b = wg >> 10, g = wg & 1023;
  long ybase = (long)b * CM * NN + (long)g * 64;
  long obase = (long)b * CC * NN + (long)g * 64;
  int t = threadIdx.x;

  const float4* w4 = (const float4*)w2;
  for (int ch = t; ch < 4096; ch += 256) {
    float4 v = w4[ch];
    int m = ch >> 6, k = (ch & 63) * 4;
    int kp = kperm(k);
    sW[m][kp + 0] = (_Float16)v.x;
    sW[m][kp + 1] = (_Float16)v.y;
    sW[m][kp + 2] = (_Float16)v.z;
    sW[m][kp + 3] = (_Float16)v.w;
  }
  for (int ch = t; ch < 2048; ch += 256) {
    int k = ch >> 3, pc = (ch & 7) * 8;
    union { uint4 q; _Float16 h[8]; } u;
    u.q = ((const uint4*)(yd + ybase + (long)k * NN))[ch & 7];
    #pragma unroll
    for (int j = 0; j < 8; ++j) sYT[pc + j][kperm(k)] = u.h[j];
  }
  __syncthreads();

  int wv = t >> 5, lane = t & 31, lm = lane & 15, hs = lane >> 4;
  int mrow = (wv >> 1) * 16 + lm;          // wave's A row
  int n0   = (wv & 1) * 32;                // wave's two n-tiles: n0, n0+16
  v8f acc0 = {}, acc1 = {};

  #pragma unroll
  for (int k0 = 0; k0 < 256; k0 += 32) {
    v16h a  = ld_frag(&sW [mrow][k0 + hs * 16]);
    v16h b0 = ld_frag(&sYT[n0 + lm][k0 + hs * 16]);
    v16h b1 = ld_frag(&sYT[n0 + 16 + lm][k0 + hs * 16]);
    acc0 = wmma_f16(a, b0, acc0);
    acc1 = wmma_f16(a, b1, acc1);
  }

  V8F r0; r0.v = acc0;
  V8F r1; r1.v = acc1;
  int m0 = (wv >> 1) * 16;
  #pragma unroll
  for (int q = 0; q < 8; ++q) {
    int mm = m0 + q + hs * 8;
    long oa = obase + (long)mm * NN + n0 + lm;
    long ob = oa + 16;
    out[oa] = r0.e[q] + x1[oa];
    out[ob] = r1.e[q] + x1[ob];
  }
}

// ---------------------------------------------------------------------------
extern "C" void kernel_launch(void* const* d_in, const int* in_sizes, int n_in,
                              void* d_out, int out_size, void* d_ws, size_t ws_size,
                              hipStream_t stream)
{
  (void)in_sizes; (void)n_in; (void)out_size; (void)ws_size;
  const float* x       = (const float*)d_in[0];
  const float* Wq      = (const float*)d_in[1];
  const float* Wk      = (const float*)d_in[2];
  const float* Wv      = (const float*)d_in[3];
  const float* rescale = (const float*)d_in[4];
  const float* Wp      = (const float*)d_in[5];
  const float* bp      = (const float*)d_in[6];
  const float* gamma   = (const float*)d_in[7];
  const float* beta    = (const float*)d_in[8];
  const float* w1      = (const float*)d_in[9];
  const float* wd      = (const float*)d_in[10];
  const float* w2      = (const float*)d_in[11];

  float* out = (float*)d_out;
  float* attn_out = out + (size_t)BB * CC * NN;     // tuple: [out | attn]

  char* ws = (char*)d_ws;
  size_t off = 0;
  float*    xn    = (float*)(ws + off);    off += (size_t)BB * CC * NN * 4;
  float*    x1    = (float*)(ws + off);    off += (size_t)BB * CC * NN * 4;
  _Float16* xn2   = (_Float16*)(ws + off); off += (size_t)BB * CC * NN * 2;
  _Float16* y1    = (_Float16*)(ws + off); off += (size_t)BB * CM * NN * 2;
  _Float16* yd    = (_Float16*)(ws + off); off += (size_t)BB * CM * NN * 2;
  float*    gpart = (float*)(ws + off);    off += (size_t)BB * 256 * 4096 * 4;
  float*    G     = (float*)(ws + off);    off += (size_t)BB * 4096 * 4;
  float*    M2    = (float*)(ws + off);    off += (size_t)BB * INNER * 64 * 4;
  float*    P     = (float*)(ws + off);    off += (size_t)BB * 4096 * 4;

  k_ln_gram    <<<BB * 256, 256, 0, stream>>>(x, gamma, beta, xn, gpart);
  k_gram_reduce<<<BB,       256, 0, stream>>>(gpart, G);
  k_attn       <<<BB * HEADS, 64, 0, stream>>>(G, Wq, Wk, Wv, rescale, attn_out, M2);
  k_pmat       <<<BB,       256, 0, stream>>>(Wp, M2, P);
  k_apply_ln2  <<<BB * 1024, 256, 0, stream>>>(x, xn, P, bp, gamma, beta, x1, xn2);
  k_ffn1       <<<BB * 1024, 256, 0, stream>>>(xn2, w1, y1);
  k_dw         <<<(BB * CM * NN) / 256, 256, 0, stream>>>(y1, wd, yd);
  k_ffn2       <<<BB * 1024, 256, 0, stream>>>(yd, w2, x1, out);
}
